// MoEGPT_58179626991690
// MI455X (gfx1250) — matrix operand order repaired and checked by reading, not compiled, not run on hardware
//
#include <hip/hip_runtime.h>

// ---------------- problem constants ----------------
#define T_TOK 8192      // B*S tokens
#define DIM   2048      // model dim D
#define HID   1024      // expert hidden H
#define NE    8         // experts
#define TOPK  2
#define AUXC  0.01f
#define ZLC   0.001f

typedef __bf16 bf16_t;
typedef __attribute__((ext_vector_type(16))) __bf16 v16bf;
typedef __attribute__((ext_vector_type(8)))  float  v8f;

struct QPair { uint4 a, b; };
union  FragU { v16bf v; QPair q; };

// LDS tile row stride: 40 bf16 = 80 bytes (20 banks -> 16 consecutive rows conflict-free)
#define LROW 40

// A fragment (16x32 bf16): lanes 0-15 row M=lane, K = {half*8..+7} U {16+half*8..+7}
__device__ __forceinline__ v16bf ld_fragA(const bf16_t* lds, int row0, int lane) {
  const char* rp = (const char*)(lds + (size_t)(row0 + (lane & 15)) * LROW);
  const int hb = ((lane >> 4) & 1) * 16;           // byte offset of K=half*8
  FragU f;
  f.q.a = *(const uint4*)(rp + hb);
  f.q.b = *(const uint4*)(rp + 32 + hb);
  return f.v;
}
// B fragment (32x16 bf16): lane col N=lane&15, K = {half*8..+7} U {16+half*8..+7}
__device__ __forceinline__ v16bf ld_fragB(const bf16_t* lds, int col0, int lane) {
  const char* rp = (const char*)(lds + (size_t)(col0 + (lane & 15)) * LROW);
  const int hb = ((lane >> 4) & 1) * 16;
  FragU f;
  f.q.a = *(const uint4*)(rp + hb);
  f.q.b = *(const uint4*)(rp + hb + 32);
  return f.v;
}
// fp32x4 -> bf16x4 packed 8-byte LDS store
__device__ __forceinline__ void pack4(bf16_t* dst, float4 f) {
  union { bf16_t h[4]; unsigned long long u; } p;
  p.h[0] = (bf16_t)f.x; p.h[1] = (bf16_t)f.y; p.h[2] = (bf16_t)f.z; p.h[3] = (bf16_t)f.w;
  *(unsigned long long*)dst = p.u;
}

// ---------------- kernel 0: zero init ----------------
__global__ void k_zero(float* __restrict__ out, size_t n,
                       int* __restrict__ cnt, float* __restrict__ sumP,
                       float* __restrict__ sumL) {
  size_t i = (size_t)blockIdx.x * blockDim.x + threadIdx.x;
  const size_t stride = (size_t)gridDim.x * blockDim.x;
  for (; i < n; i += stride) out[i] = 0.0f;
  if (blockIdx.x == 0 && threadIdx.x < NE) {
    cnt[threadIdx.x] = 0;
    sumP[threadIdx.x] = 0.0f;
    if (threadIdx.x == 0) sumL[0] = 0.0f;
  }
}

// ---------------- kernel 1: router (one wave32 per token) ----------------
__global__ __launch_bounds__(256) void k_router(
    const float* __restrict__ x, const float* __restrict__ Wr,
    int* __restrict__ cnt, int* __restrict__ idxL, float* __restrict__ wL,
    float* __restrict__ sumP, float* __restrict__ sumL) {
  const int lane = threadIdx.x & 31;
  const int wid  = threadIdx.x >> 5;
  const int t    = blockIdx.x * 8 + wid;          // 1024 blocks * 8 waves = 8192 tokens
  const float* xr = x + (size_t)t * DIM;

  float acc[NE];
#pragma unroll
  for (int e = 0; e < NE; ++e) acc[e] = 0.0f;

  for (int d = lane; d < DIM; d += 32) {
    const float xv = xr[d];
#pragma unroll
    for (int e = 0; e < NE; ++e) acc[e] += xv * Wr[e * DIM + d];
  }
#pragma unroll
  for (int off = 16; off > 0; off >>= 1) {
#pragma unroll
    for (int e = 0; e < NE; ++e) acc[e] += __shfl_xor(acc[e], off, 32);
  }

  if (lane == 0) {
    float m = acc[0];
#pragma unroll
    for (int e = 1; e < NE; ++e) m = fmaxf(m, acc[e]);
    float ex[NE], s = 0.0f;
#pragma unroll
    for (int e = 0; e < NE; ++e) { ex[e] = __expf(acc[e] - m); s += ex[e]; }
    const float inv = 1.0f / s;
    float p[NE];
#pragma unroll
    for (int e = 0; e < NE; ++e) p[e] = ex[e] * inv;
    const float lse = m + __logf(s);

    int e0 = 0;
#pragma unroll
    for (int e = 1; e < NE; ++e) if (p[e] > p[e0]) e0 = e;
    int e1 = (e0 == 0) ? 1 : 0;
#pragma unroll
    for (int e = 0; e < NE; ++e) if (e != e0 && p[e] > p[e1]) e1 = e;

    const float den = p[e0] + p[e1] + 1e-8f;
    const float w0 = p[e0] / den, w1 = p[e1] / den;

    int p0 = atomicAdd(&cnt[e0], 1);
    idxL[e0 * T_TOK + p0] = t;  wL[e0 * T_TOK + p0] = w0;
    int p1 = atomicAdd(&cnt[e1], 1);
    idxL[e1 * T_TOK + p1] = t;  wL[e1 * T_TOK + p1] = w1;

#pragma unroll
    for (int e = 0; e < NE; ++e) atomicAdd(&sumP[e], p[e]);
    atomicAdd(sumL, lse * lse);
  }
}

// ---------------- kernel 2: scan counts + aux loss ----------------
__global__ void k_scan_aux(const int* __restrict__ cnt, int* __restrict__ offs,
                           const float* __restrict__ sumP, const float* __restrict__ sumL,
                           float* __restrict__ auxOut) {
  if (threadIdx.x == 0) {
    int run = 0; float bal = 0.0f;
    for (int e = 0; e < NE; ++e) {
      offs[e] = run; run += cnt[e];
      bal += ((float)cnt[e] / (float)T_TOK) * (sumP[e] / (float)T_TOK);
    }
    auxOut[0] = AUXC * (float)NE * bal + ZLC * (sumL[0] / (float)T_TOK);
  }
}

// ---------------- kernel 3: expert FC  h = relu(x_g @ Wfc^T)^2 * cw  (bf16 out) ----------------
__global__ __launch_bounds__(256) void k_moe_fc(
    const float* __restrict__ x, const float* __restrict__ Wfc,
    const int* __restrict__ cnt, const int* __restrict__ offs,
    const int* __restrict__ idxL, const float* __restrict__ wL,
    bf16_t* __restrict__ hbuf) {
  const int e     = blockIdx.z;
  const int count = cnt[e];
  const int mBase = blockIdx.x * 128;
  if (mBase >= count) return;
  const int nBase = blockIdx.y * 128;                    // over H
  const int hoff  = offs[e];
  const float* W  = Wfc + (size_t)e * HID * DIM;

  __shared__ alignas(16) bf16_t lA[128 * LROW];
  __shared__ alignas(16) bf16_t lB[128 * LROW];
  __shared__ int   idxT[128];
  __shared__ float gwT[128];

  const int tid  = threadIdx.x;
  const int lane = tid & 31;
  const int wid  = tid >> 5;
  const int wM   = (wid & 3) * 32;                       // row group
  const int wN   = (wid >> 2) * 64;                      // col group

  if (tid < 128) {
    const int mg = mBase + tid;
    if (mg < count) { idxT[tid] = idxL[e * T_TOK + mg]; gwT[tid] = wL[e * T_TOK + mg]; }
    else            { idxT[tid] = -1;                    gwT[tid] = 0.0f; }
  }

  v8f accv[2][4];
#pragma unroll
  for (int i = 0; i < 2; ++i)
#pragma unroll
    for (int j = 0; j < 4; ++j)
#pragma unroll
      for (int k = 0; k < 8; ++k) accv[i][j][k] = 0.0f;

  for (int kt = 0; kt < DIM; kt += 32) {
    __syncthreads();
    // A: gathered x rows, fp32 -> bf16  (128 x 32)
#pragma unroll
    for (int l = 0; l < 4; ++l) {
      const int lin = l * 256 + tid;
      const int row = lin >> 3, c4 = lin & 7;
      const int tok = idxT[row];
      float4 f = make_float4(0.f, 0.f, 0.f, 0.f);
      if (tok >= 0) f = *(const float4*)(x + (size_t)tok * DIM + kt + c4 * 4);
      pack4(lA + row * LROW + c4 * 4, f);
    }
    // B: Wfc rows [nBase..+127], fp32 -> bf16
#pragma unroll
    for (int l = 0; l < 4; ++l) {
      const int lin = l * 256 + tid;
      const int row = lin >> 3, c4 = lin & 7;
      float4 f = *(const float4*)(W + (size_t)(nBase + row) * DIM + kt + c4 * 4);
      pack4(lB + row * LROW + c4 * 4, f);
    }
    __syncthreads();

    v16bf aF[2], bF[4];
#pragma unroll
    for (int i = 0; i < 2; ++i) aF[i] = ld_fragA(lA, wM + i * 16, lane);
#pragma unroll
    for (int j = 0; j < 4; ++j) bF[j] = ld_fragB(lB, wN + j * 16, lane);
#pragma unroll
    for (int i = 0; i < 2; ++i)
#pragma unroll
      for (int j = 0; j < 4; ++j)
        accv[i][j] = __builtin_amdgcn_wmma_f32_16x16x32_bf16(
            false, aF[i], false, bF[j], (short)0, accv[i][j], false, false);
  }

  const int nl = lane & 15, half = (lane >> 4) & 1;
#pragma unroll
  for (int i = 0; i < 2; ++i)
#pragma unroll
    for (int j = 0; j < 4; ++j)
#pragma unroll
      for (int r = 0; r < 8; ++r) {
        const int ml = wM + i * 16 + half * 8 + r;
        const int mG = mBase + ml;
        if (mG < count) {
          float v = accv[i][j][r];
          v = fmaxf(v, 0.0f);
          v = v * v * gwT[ml];
          const int nG = nBase + wN + j * 16 + nl;
          hbuf[(size_t)(hoff + mG) * HID + nG] = (bf16_t)v;
        }
      }
}

// ---------------- kernel 4: expert PROJ  out[t] += h2w @ Wproj^T (atomic scatter) ----------------
__global__ __launch_bounds__(256) void k_moe_proj(
    const bf16_t* __restrict__ hbuf, const float* __restrict__ Wproj,
    const int* __restrict__ cnt, const int* __restrict__ offs,
    const int* __restrict__ idxL, float* __restrict__ out) {
  const int e     = blockIdx.z;
  const int count = cnt[e];
  const int mBase = blockIdx.x * 128;
  if (mBase >= count) return;
  const int nBase = blockIdx.y * 128;                    // over D
  const int hoff  = offs[e];
  const float* W  = Wproj + (size_t)e * DIM * HID;

  __shared__ alignas(16) bf16_t lA[128 * LROW];
  __shared__ alignas(16) bf16_t lB[128 * LROW];
  __shared__ int idxT[128];

  const int tid  = threadIdx.x;
  const int lane = tid & 31;
  const int wid  = tid >> 5;
  const int wM   = (wid & 3) * 32;
  const int wN   = (wid >> 2) * 64;

  if (tid < 128) {
    const int mg = mBase + tid;
    idxT[tid] = (mg < count) ? idxL[e * T_TOK + mg] : -1;
  }

  v8f accv[2][4];
#pragma unroll
  for (int i = 0; i < 2; ++i)
#pragma unroll
    for (int j = 0; j < 4; ++j)
#pragma unroll
      for (int k = 0; k < 8; ++k) accv[i][j][k] = 0.0f;

  for (int kt = 0; kt < HID; kt += 32) {
    __syncthreads();
    // A: compact h rows, already bf16  (128 x 32 -> 512 x 16B chunks)
#pragma unroll
    for (int l = 0; l < 2; ++l) {
      const int lin = l * 256 + tid;
      const int row = lin >> 2, c = lin & 3;
      const int mg  = mBase + row;
      uint4 q = make_uint4(0u, 0u, 0u, 0u);
      if (mg < count)
        q = *(const uint4*)(hbuf + (size_t)(hoff + mg) * HID + kt + c * 8);
      *(uint4*)((char*)lA + row * (LROW * 2) + c * 16) = q;
    }
    // B: Wproj rows [nBase..+127], fp32 -> bf16
#pragma unroll
    for (int l = 0; l < 4; ++l) {
      const int lin = l * 256 + tid;
      const int row = lin >> 3, c4 = lin & 7;
      float4 f = *(const float4*)(W + (size_t)(nBase + row) * HID + kt + c4 * 4);
      pack4(lB + row * LROW + c4 * 4, f);
    }
    __syncthreads();

    v16bf aF[2], bF[4];
#pragma unroll
    for (int i = 0; i < 2; ++i) aF[i] = ld_fragA(lA, wM + i * 16, lane);
#pragma unroll
    for (int j = 0; j < 4; ++j) bF[j] = ld_fragB(lB, wN + j * 16, lane);
#pragma unroll
    for (int i = 0; i < 2; ++i)
#pragma unroll
      for (int j = 0; j < 4; ++j)
        accv[i][j] = __builtin_amdgcn_wmma_f32_16x16x32_bf16(
            false, aF[i], false, bF[j], (short)0, accv[i][j], false, false);
  }

  const int nl = lane & 15, half = (lane >> 4) & 1;
#pragma unroll
  for (int i = 0; i < 2; ++i)
#pragma unroll
    for (int j = 0; j < 4; ++j)
#pragma unroll
      for (int r = 0; r < 8; ++r) {
        const int ml = wM + i * 16 + half * 8 + r;
        const int mG = mBase + ml;
        if (mG < count) {
          const int tok = idxT[ml];
          const int nG  = nBase + wN + j * 16 + nl;
          atomicAdd(out + (size_t)tok * DIM + nG, accv[i][j][r]);
        }
      }
}

// ---------------- host launcher ----------------
extern "C" void kernel_launch(void* const* d_in, const int* in_sizes, int n_in,
                              void* d_out, int out_size, void* d_ws, size_t ws_size,
                              hipStream_t stream) {
  (void)in_sizes; (void)n_in; (void)out_size; (void)ws_size;
  const float* x     = (const float*)d_in[0];   // [T, D]
  const float* Wr    = (const float*)d_in[1];   // [E, D]
  const float* Wfc   = (const float*)d_in[2];   // [E, H, D]
  const float* Wproj = (const float*)d_in[3];   // [E, D, H]
  float* out = (float*)d_out;                   // [T*D] + aux scalar

  char* ws = (char*)d_ws;
  size_t off = 0;
  bf16_t* hbuf = (bf16_t*)(ws + off); off += (size_t)T_TOK * TOPK * HID * sizeof(bf16_t); // 33.5 MB
  int*    cnt  = (int*)(ws + off);    off += 64;
  int*    offs = (int*)(ws + off);    off += 64;
  float*  sumP = (float*)(ws + off);  off += 64;
  float*  sumL = (float*)(ws + off);  off += 64;
  int*    idxL = (int*)(ws + off);    off += (size_t)NE * T_TOK * sizeof(int);
  float*  wLst = (float*)(ws + off);  off += (size_t)NE * T_TOK * sizeof(float);

  const size_t nOut = (size_t)T_TOK * DIM + 1;
  k_zero<<<4096, 256, 0, stream>>>(out, nOut, cnt, sumP, sumL);
  k_router<<<T_TOK / 8, 256, 0, stream>>>(x, Wr, cnt, idxL, wLst, sumP, sumL);
  k_scan_aux<<<1, 32, 0, stream>>>(cnt, offs, sumP, sumL, out + (size_t)T_TOK * DIM);
  k_moe_fc<<<dim3(T_TOK / 128, HID / 128, NE), 256, 0, stream>>>(
      x, Wfc, cnt, offs, idxL, wLst, hbuf);
  k_moe_proj<<<dim3(T_TOK / 128, DIM / 128, NE), 256, 0, stream>>>(
      hbuf, Wproj, cnt, offs, idxL, out);
}
